// MambaBlock_56942676411176
// MI455X (gfx1250) — compile-verified, matrix operand order checked by reading, not run on hardware
//
#include <hip/hip_runtime.h>
#include <math.h>

// ---------------------------------------------------------------------------
// Mamba block for MI455X (gfx1250, wave32, WMMA).
// All GEMMs consume pre-converted bf16 operands -> v_wmma_f32_16x16x32_bf16
// with zero converts in the K loop; 8 independent WMMA chains per wave.
// ---------------------------------------------------------------------------

#define D_MODEL 1024
#define D_STATE 16
#define D_CONV 4
#define D_INNER 2048
#define DT_RANK 64
#define BATCH 2
#define SEQ 1024
#define MROWS (BATCH * SEQ)   // 2048; multiple of 64 -> no M bounds checks

typedef __bf16 bf16x16 __attribute__((ext_vector_type(16)));
typedef float  v8f     __attribute__((ext_vector_type(8)));

#define ACT_NONE     0
#define ACT_SOFTPLUS 1
#define ACT_GELU     2

// ---------------------------------------------------------------------------
// f32 -> bf16 elementwise cast (vectorized x4). n is a multiple of 4 here.
// ---------------------------------------------------------------------------
__global__ __launch_bounds__(256)
void cast_f32_bf16(const float* __restrict__ in, __bf16* __restrict__ out, int n)
{
    int i = (blockIdx.x * 256 + threadIdx.x) * 4;
    if (i + 3 < n) {
        float4 v = *(const float4*)(in + i);
        out[i + 0] = (__bf16)v.x;
        out[i + 1] = (__bf16)v.y;
        out[i + 2] = (__bf16)v.z;
        out[i + 3] = (__bf16)v.w;
    } else {
        for (; i < n; ++i) out[i] = (__bf16)in[i];
    }
}

// ---------------------------------------------------------------------------
// WMMA GEMM:  C[M,N] = act( A[M,K](bf16) * W[N,K]^T(bf16) + bias ) (+ resid)
// Block: 256 threads = 8 waves arranged 2(M) x 4(N).
// Wave tile: 32 x 64 (2 A-frags x 4 B-frags = 8 accumulators).
// Block tile: 64 x 256.   Grid: ( ceil(N/256), M/64 ).
// C (f32) and/or Cb (bf16) outputs, both row stride ldc.
// ---------------------------------------------------------------------------
__global__ __launch_bounds__(256)
void wmma_gemm_bf16(const __bf16* __restrict__ A,
                    const __bf16* __restrict__ W,
                    const float* __restrict__ bias,   // N elems or nullptr
                    const float* __restrict__ resid,  // M x ldc or nullptr
                    float* __restrict__ C,            // nullable
                    __bf16* __restrict__ Cb,          // nullable
                    int N, int K, int lda, int ldc, int act)
{
    const int wave   = threadIdx.x >> 5;
    const int lane   = threadIdx.x & 31;
    const int wm     = wave >> 2;            // 0..1
    const int wn     = wave & 3;             // 0..3
    const int lane15 = lane & 15;
    const int kHalf  = (lane >> 4) * 16;     // upper K half for lanes 16..31

    const int rowBase = blockIdx.y * 64 + wm * 32;
    const int colBase = blockIdx.x * 256 + wn * 64;

    const size_t aOff0 = (size_t)(rowBase + lane15) * lda;
    const size_t aOff1 = aOff0 + (size_t)16 * lda;

    int ccol[4];
    #pragma unroll
    for (int t = 0; t < 4; ++t) {
        int c = colBase + t * 16 + lane15;
        ccol[t] = (c < N) ? c : (N - 1);     // clamp OOB weight loads
    }

    v8f acc[2][4] = {};

    for (int k0 = 0; k0 < K; k0 += 32) {
        const bf16x16 a0 = *(const bf16x16*)(A + aOff0 + k0 + kHalf);
        const bf16x16 a1 = *(const bf16x16*)(A + aOff1 + k0 + kHalf);
        #pragma unroll
        for (int t = 0; t < 4; ++t) {
            const bf16x16 b = *(const bf16x16*)(W + (size_t)ccol[t] * K + k0 + kHalf);
            acc[0][t] = __builtin_amdgcn_wmma_f32_16x16x32_bf16(
                            false, a0, false, b, (short)0, acc[0][t], false, false);
            acc[1][t] = __builtin_amdgcn_wmma_f32_16x16x32_bf16(
                            false, a1, false, b, (short)0, acc[1][t], false, false);
        }
    }

    // C/D layout: VGPR i -> row = i (+8 for lanes 16..31), col = lane&15
    const int rOfs = (lane >> 4) * 8;
    #pragma unroll
    for (int mi = 0; mi < 2; ++mi) {
        #pragma unroll
        for (int t = 0; t < 4; ++t) {
            const int col = colBase + t * 16 + lane15;
            if (col >= N) continue;
            #pragma unroll
            for (int i = 0; i < 8; ++i) {
                const int row = rowBase + mi * 16 + i + rOfs;
                float v = acc[mi][t][i];
                if (bias) v += bias[col];
                if (act == ACT_SOFTPLUS) {
                    v = (v > 20.0f) ? v : log1pf(__expf(v));
                } else if (act == ACT_GELU) {
                    v = 0.5f * v * (1.0f + erff(v * 0.70710678118654752f));
                }
                const size_t idx = (size_t)row * ldc + col;
                if (resid) v += resid[idx];
                if (C)  C[idx]  = v;
                if (Cb) Cb[idx] = (__bf16)v;
            }
        }
    }
}

// ---------------------------------------------------------------------------
// RMSNorm per the reference: out = w * x / sqrt(sum(x*x) + eps)  (sum, not mean)
// One block (256 threads) per row of 1024; bf16 output (feeds a GEMM).
// ---------------------------------------------------------------------------
__global__ __launch_bounds__(256)
void rmsnorm_bf16_kernel(const float* __restrict__ x, const float* __restrict__ w,
                         __bf16* __restrict__ out)
{
    const int row = blockIdx.x;
    const float* xr = x + (size_t)row * D_MODEL;
    float s = 0.0f;
    for (int c = threadIdx.x; c < D_MODEL; c += 256) { float v = xr[c]; s += v * v; }
    __shared__ float red[256];
    red[threadIdx.x] = s;
    __syncthreads();
    for (int o = 128; o > 0; o >>= 1) {
        if (threadIdx.x < o) red[threadIdx.x] += red[threadIdx.x + o];
        __syncthreads();
    }
    const float inv = rsqrtf(red[0] + 1e-6f);
    __bf16* orow = out + (size_t)row * D_MODEL;
    for (int c = threadIdx.x; c < D_MODEL; c += 256)
        orow[c] = (__bf16)(w[c] * xr[c] * inv);
}

// ---------------------------------------------------------------------------
// Depthwise causal conv (D_CONV=4) + SiLU.  xm = xz[:, :, 0:2048] (stride 4096)
// Writes u (f32, for the scan) and ub (bf16, for the x_proj GEMM).
// ---------------------------------------------------------------------------
__global__ __launch_bounds__(256)
void conv_silu_kernel(const float* __restrict__ xz, const float* __restrict__ cw,
                      const float* __restrict__ cb,
                      float* __restrict__ u, __bf16* __restrict__ ub)
{
    const int idx = blockIdx.x * 256 + threadIdx.x;   // over B*L*D_INNER
    const int d  = idx & (D_INNER - 1);
    const int bl = idx >> 11;
    const int l  = bl & (SEQ - 1);
    const int b  = bl >> 10;
    float acc = cb[d];
    #pragma unroll
    for (int j = 0; j < D_CONV; ++j) {
        int lj = l - (D_CONV - 1) + j;
        if (lj >= 0)
            acc += cw[d * D_CONV + j] * xz[((size_t)(b * SEQ + lj)) * (2 * D_INNER) + d];
    }
    const float s = acc / (1.0f + __expf(-acc));
    u[(size_t)bl * D_INNER + d]  = s;
    ub[(size_t)bl * D_INNER + d] = (__bf16)s;
}

// ---------------------------------------------------------------------------
// Selective scan fused with D-skip and SiLU(z) gating; bf16 output (-> out_proj).
// One thread per channel; h[16] in registers; B_t/C_t broadcast via LDS.
// Grid: (D_INNER/256, BATCH).
// ---------------------------------------------------------------------------
__global__ __launch_bounds__(256)
void scan_kernel(const float* __restrict__ delta, const float* __restrict__ u,
                 const float* __restrict__ dbc,   const float* __restrict__ xz,
                 const float* __restrict__ A_log, const float* __restrict__ Dp,
                 __bf16* __restrict__ yg)
{
    const int b  = blockIdx.y;
    const int dg = blockIdx.x * 256 + threadIdx.x;

    float Ad[D_STATE], h[D_STATE];
    #pragma unroll
    for (int n = 0; n < D_STATE; ++n) {
        Ad[n] = -__expf(A_log[(size_t)dg * D_STATE + n]);
        h[n] = 0.0f;
    }
    const float Dd = Dp[dg];

    __shared__ float Bs[D_STATE];
    __shared__ float Cs[D_STATE];

    for (int t = 0; t < SEQ; ++t) {
        const int row = b * SEQ + t;
        if (threadIdx.x < 32) {
            float v = dbc[(size_t)row * (DT_RANK + 2 * D_STATE) + DT_RANK + threadIdx.x];
            if (threadIdx.x < D_STATE) Bs[threadIdx.x] = v;
            else                       Cs[threadIdx.x - D_STATE] = v;
        }
        __syncthreads();

        const float dt = delta[(size_t)row * D_INNER + dg];
        const float ut = u[(size_t)row * D_INNER + dg];
        float y = 0.0f;
        #pragma unroll
        for (int n = 0; n < D_STATE; ++n) {
            const float dA = __expf(dt * Ad[n]);
            h[n] = dA * h[n] + dt * Bs[n] * ut;
            y += h[n] * Cs[n];
        }
        const float z  = xz[(size_t)row * (2 * D_INNER) + D_INNER + dg];
        const float sz = z / (1.0f + __expf(-z));
        yg[(size_t)row * D_INNER + dg] = (__bf16)((y + ut * Dd) * sz);
        __syncthreads();
    }
}

// ---------------------------------------------------------------------------
// Host-side launch
// ---------------------------------------------------------------------------
static inline dim3 gemm_grid(int N) { return dim3((N + 255) / 256, MROWS / 64); }

extern "C" void kernel_launch(void* const* d_in, const int* in_sizes, int n_in,
                              void* d_out, int out_size, void* d_ws, size_t ws_size,
                              hipStream_t stream)
{
    (void)in_sizes; (void)n_in; (void)out_size; (void)ws_size;

    const float* x         = (const float*)d_in[0];
    const float* norm1_w   = (const float*)d_in[1];
    const float* norm2_w   = (const float*)d_in[2];
    const float* in_proj_w = (const float*)d_in[3];
    const float* conv_w    = (const float*)d_in[4];
    const float* conv_b    = (const float*)d_in[5];
    const float* x_proj_w  = (const float*)d_in[6];
    const float* dt_proj_w = (const float*)d_in[7];
    const float* dt_proj_b = (const float*)d_in[8];
    const float* A_log     = (const float*)d_in[9];
    const float* Dp        = (const float*)d_in[10];
    const float* out_proj_w= (const float*)d_in[11];
    const float* mlp_w1    = (const float*)d_in[12];
    const float* mlp_b1    = (const float*)d_in[13];
    const float* mlp_w2    = (const float*)d_in[14];
    const float* mlp_b2    = (const float*)d_in[15];
    float* out = (float*)d_out;

    // ---- workspace layout (256B-aligned carve-out) ----
    char* cur = (char*)d_ws;
    auto carve = [&](size_t bytes) {
        char* p = cur;
        cur += (bytes + 255) & ~(size_t)255;
        return p;
    };
    __bf16* wb_in  = (__bf16*)carve((size_t)(2*D_INNER)*D_MODEL * 2);   // 4096x1024
    __bf16* wb_xp  = (__bf16*)carve((size_t)96*D_INNER * 2);            // 96x2048
    __bf16* wb_dt  = (__bf16*)carve((size_t)D_INNER*DT_RANK * 2);       // 2048x64
    __bf16* wb_out = (__bf16*)carve((size_t)D_MODEL*D_INNER * 2);       // 1024x2048
    __bf16* wb_m1  = (__bf16*)carve((size_t)(4*D_MODEL)*D_MODEL * 2);   // 4096x1024
    __bf16* wb_m2  = (__bf16*)carve((size_t)D_MODEL*(4*D_MODEL) * 2);   // 1024x4096
    __bf16* xnb    = (__bf16*)carve((size_t)MROWS*D_MODEL * 2);
    float*  xz     = (float*) carve((size_t)MROWS*(2*D_INNER) * 4);     // reused as m1
    float*  u      = (float*) carve((size_t)MROWS*D_INNER * 4);
    __bf16* ub     = (__bf16*)carve((size_t)MROWS*D_INNER * 2);
    float*  dbc    = (float*) carve((size_t)MROWS*96 * 4);
    __bf16* dbcb   = (__bf16*)carve((size_t)MROWS*96 * 2);
    float*  delta  = (float*) carve((size_t)MROWS*D_INNER * 4);
    __bf16* ygb    = (__bf16*)carve((size_t)MROWS*D_INNER * 2);
    float*  h      = (float*) carve((size_t)MROWS*D_MODEL * 4);
    __bf16* hnb    = (__bf16*)carve((size_t)MROWS*D_MODEL * 2);
    __bf16* m1     = (__bf16*)xz;   // mlp hidden reuses xz (dead after the scan)

    const dim3 blk(256);
    auto castN = [&](const float* src, __bf16* dst, int n) {
        cast_f32_bf16<<<(n / 4 + 255) / 256, blk, 0, stream>>>(src, dst, n);
    };

    // 0) weights -> bf16 (once per call; ~15M elems, negligible vs 23.3 TB/s)
    castN(in_proj_w,  wb_in,  2*D_INNER*D_MODEL);
    castN(x_proj_w,   wb_xp,  96*D_INNER);
    castN(dt_proj_w,  wb_dt,  D_INNER*DT_RANK);
    castN(out_proj_w, wb_out, D_MODEL*D_INNER);
    castN(mlp_w1,     wb_m1,  4*D_MODEL*D_MODEL);
    castN(mlp_w2,     wb_m2,  D_MODEL*4*D_MODEL);

    // 1) RMSNorm(x) -> xnb (bf16)
    rmsnorm_bf16_kernel<<<MROWS, blk, 0, stream>>>(x, norm1_w, xnb);

    // 2) in_proj: xz = xnb @ wb_in^T          (N=4096, K=1024) -> f32
    wmma_gemm_bf16<<<gemm_grid(2*D_INNER), blk, 0, stream>>>(
        xnb, wb_in, nullptr, nullptr, xz, nullptr,
        2*D_INNER, D_MODEL, D_MODEL, 2*D_INNER, ACT_NONE);

    // 3) depthwise conv + SiLU -> u (f32) + ub (bf16)
    conv_silu_kernel<<<(MROWS * D_INNER) / 256, blk, 0, stream>>>(xz, conv_w, conv_b, u, ub);

    // 4) x_proj: dbc = ub @ wb_xp^T           (N=96, K=2048) -> f32 + bf16
    wmma_gemm_bf16<<<gemm_grid(96), blk, 0, stream>>>(
        ub, wb_xp, nullptr, nullptr, dbc, dbcb,
        96, D_INNER, D_INNER, 96, ACT_NONE);

    // 5) dt_proj + softplus: delta            (N=2048, K=64, A=dbcb[:, :64], lda=96)
    wmma_gemm_bf16<<<gemm_grid(D_INNER), blk, 0, stream>>>(
        dbcb, wb_dt, dt_proj_b, nullptr, delta, nullptr,
        D_INNER, DT_RANK, 96, D_INNER, ACT_SOFTPLUS);

    // 6) selective scan + D-skip + SiLU(z) gating -> ygb (bf16)
    scan_kernel<<<dim3(D_INNER / 256, BATCH), blk, 0, stream>>>(
        delta, u, dbc, xz, A_log, Dp, ygb);

    // 7) out_proj + residual(x): h            (N=1024, K=2048) -> f32
    wmma_gemm_bf16<<<gemm_grid(D_MODEL), blk, 0, stream>>>(
        ygb, wb_out, nullptr, x, h, nullptr,
        D_MODEL, D_INNER, D_INNER, D_MODEL, ACT_NONE);

    // 8) RMSNorm(h) -> hnb (bf16)
    rmsnorm_bf16_kernel<<<MROWS, blk, 0, stream>>>(h, norm2_w, hnb);

    // 9) mlp1 + bias + exact GELU -> m1 (bf16 only)   (N=4096, K=1024)
    wmma_gemm_bf16<<<gemm_grid(4*D_MODEL), blk, 0, stream>>>(
        hnb, wb_m1, mlp_b1, nullptr, nullptr, m1,
        4*D_MODEL, D_MODEL, D_MODEL, 4*D_MODEL, ACT_GELU);

    // 10) mlp2 + bias + residual(h) -> out (f32)      (N=1024, K=4096)
    wmma_gemm_bf16<<<gemm_grid(D_MODEL), blk, 0, stream>>>(
        m1, wb_m2, mlp_b2, h, out, nullptr,
        D_MODEL, 4*D_MODEL, 4*D_MODEL, D_MODEL, ACT_NONE);
}
